// EncoderLayer_23081154249352
// MI455X (gfx1250) — compile-verified, matrix operand order checked by reading, not run on hardware
//
#include <hip/hip_runtime.h>
#include <math.h>

#define B_  2
#define N_  2048
#define K_  48
#define C_  128
#define H_  128
#define E_  384
#define EPS_ 1e-5f

typedef __attribute__((ext_vector_type(16))) _Float16 v16h;
typedef __attribute__((ext_vector_type(8)))  _Float16 v8h;
typedef __attribute__((ext_vector_type(8)))  float    v8f;

__device__ __forceinline__ float gelu_f(float x) {
  return 0.5f * x * (1.0f + erff(x * 0.7071067811865475f));
}

// ---- WMMA 16x16x32 f16->f32 fragment helpers (wave32) ----
// A (16x32 f16), per ISA 7.12.2: lane&15 = M row; element e holds
//   K = (e>=8 ? 16:0) + (lane>=16 ? 8:0) + (e&7)
__device__ __forceinline__ v16h load_a_lds(const _Float16* base, int rowStride,
                                           int mtile, int kbase, int lane) {
  const int m  = mtile * 16 + (lane & 15);
  const int kh = (lane >> 4) * 8;
  const _Float16* p = base + m * rowStride + kbase + kh;   // 16B aligned
  v8h lo = *(const v8h*)(p);
  v8h hi = *(const v8h*)(p + 16);
  v16h a;
#pragma unroll
  for (int e = 0; e < 8; ++e) { a[e] = lo[e]; a[e + 8] = hi[e]; }
  return a;
}

// B (32x16 f16) from pre-converted f16 row-major weight W[out][in]:
// B[k][n] = W[ntile*16+n][k]; lane&15 = N col, same K element mapping as A.
__device__ __forceinline__ v16h load_b_h(const _Float16* __restrict__ W, int in_f,
                                         int ntile, int kbase, int lane) {
  const int o  = ntile * 16 + (lane & 15);
  const int kh = (lane >> 4) * 8;
  const _Float16* p = W + o * in_f + kbase + kh;           // 16B aligned
  v8h lo = *(const v8h*)(p);
  v8h hi = *(const v8h*)(p + 16);
  v16h b;
#pragma unroll
  for (int e = 0; e < 8; ++e) { b[e] = lo[e]; b[e + 8] = hi[e]; }
  return b;
}

// 48xin_f (LDS f16) @ W^T -> three 16x16 f32 accumulators for this wave's ntile
__device__ __forceinline__ void gemm3(const _Float16* shIn, int inStride, int in_f,
                                      const _Float16* __restrict__ W, int nt,
                                      int lane, v8f acc[3]) {
  v8f z = {};
  acc[0] = z; acc[1] = z; acc[2] = z;
  const int nkt = in_f >> 5;
  for (int kt = 0; kt < nkt; ++kt) {
    v16h bf = load_b_h(W, in_f, nt, kt * 32, lane);
#pragma unroll
    for (int mt = 0; mt < 3; ++mt) {
      v16h af = load_a_lds(shIn, inStride, mt, kt * 32, lane);
      acc[mt] = __builtin_amdgcn_wmma_f32_16x16x32_f16(
          false, af, false, bf, (short)0, acc[mt], false, false);
    }
  }
}

__device__ __forceinline__ void gelu_store(const v8f acc[3], float bias,
                                           _Float16* shOut, int outStride,
                                           int col, int rbase) {
#pragma unroll
  for (int mt = 0; mt < 3; ++mt)
#pragma unroll
    for (int v = 0; v < 8; ++v) {
      const int row = mt * 16 + rbase + v;
      shOut[row * outStride + col] = (_Float16)gelu_f(acc[mt][v] + bias);
    }
}

// f32 -> f16 weight pre-conversion (weights are L2-resident; done once per launch)
__global__ __launch_bounds__(256, 4) void cvt_w_kernel(
    const float* __restrict__ src, _Float16* __restrict__ dst, int n) {
  int i = blockIdx.x * 256 + threadIdx.x;
  if (i < n) dst[i] = (_Float16)src[i];
}

// Shared layout (bytes), 63936 total:
//  [0,37632)      h_EV f16 [48][392]  (MODE1 epilogue: reused as m f32 [48][128])
//  [37632,50688)  act1 f16 [48][136]
//  [50688,63744)  act2 f16 [48][136]
//  [63744,63936)  mask_attend f32 [48]
template<int MODE>
__global__ __launch_bounds__(256, 1) void edge_mlp_kernel(
    const float* __restrict__ hV, const float* __restrict__ hE,
    const int* __restrict__ nIdx,
    const _Float16* __restrict__ W1, const float* __restrict__ b1,
    const _Float16* __restrict__ W2, const float* __restrict__ b2,
    const _Float16* __restrict__ W3, const float* __restrict__ b3,
    const float* __restrict__ maskAtt,
    const float* __restrict__ lnG, const float* __restrict__ lnB,
    float* __restrict__ out)
{
  __shared__ alignas(16) unsigned char smem[63936];
  _Float16* shEV = (_Float16*)smem;              // stride 392
  _Float16* shA1 = (_Float16*)(smem + 37632);    // stride 136
  _Float16* shA2 = (_Float16*)(smem + 50688);    // stride 136
  float*    shMk = (float*)(smem + 63744);
  float*    shM  = (float*)smem;                 // [48][128] f32 (MODE 1)

  const int bn   = blockIdx.x;
  const int b    = bn >> 11;                     // N_ == 2048
  const int tid  = threadIdx.x;
  const int wave = tid >> 5;
  const int lane = tid & 31;

  // ---- stage h_EV = [h_V_i | h_E | h_V_j] as f16 (gather via neighbor_idx) ----
  {
    const float* srcV = hV + bn * C_;
#pragma unroll
    for (int r = 0; r < 6; ++r) {
      const int row = wave * 6 + r;
      const int nb  = nIdx[bn * K_ + row];
      const float* srcE = hE + (bn * K_ + row) * C_;
      const float* srcN = hV + (b * N_ + nb) * C_;
      _Float16* dst = shEV + row * 392;
      for (int c = lane; c < C_; c += 32) {
        dst[c]          = (_Float16)srcV[c];
        dst[C_ + c]     = (_Float16)srcE[c];
        dst[2 * C_ + c] = (_Float16)srcN[c];
      }
    }
    if (MODE == 0) {
      for (int i = tid; i < K_; i += 256) shMk[i] = maskAtt[bn * K_ + i];
    }
  }
  __syncthreads();

  const int nt    = wave;                        // this wave's 16-col N tile
  const int col   = nt * 16 + (lane & 15);
  const int rbase = (lane >> 4) * 8;

  v8f acc[3];

  // layer 1: [48x384] @ W1^T, gelu -> act1
  gemm3(shEV, 392, E_, W1, nt, lane, acc);
  gelu_store(acc, b1[col], shA1, 136, col, rbase);
  __syncthreads();

  // layer 2: [48x128] @ W2^T, gelu -> act2
  gemm3(shA1, 136, H_, W2, nt, lane, acc);
  gelu_store(acc, b2[col], shA2, 136, col, rbase);
  __syncthreads();

  // layer 3: [48x128] @ W3^T -> m
  gemm3(shA2, 136, H_, W3, nt, lane, acc);
  const float bias3 = b3[col];

  if (MODE == 0) {
    // masked sum over K, scaled by 1/K -> dh
    float s = 0.f;
#pragma unroll
    for (int mt = 0; mt < 3; ++mt)
#pragma unroll
      for (int v = 0; v < 8; ++v) {
        const int row = mt * 16 + rbase + v;
        s += (acc[mt][v] + bias3) * shMk[row];
      }
    s += __shfl_xor(s, 16);                      // combine the two row halves
    if (lane < 16) out[bn * C_ + col] = s * (1.0f / (float)K_);
  } else {
    // residual + per-row LayerNorm: h_E = LN(h_E + m)
#pragma unroll
    for (int mt = 0; mt < 3; ++mt)
#pragma unroll
      for (int v = 0; v < 8; ++v) {
        const int row = mt * 16 + rbase + v;
        shM[row * C_ + col] = acc[mt][v] + bias3 + hE[(bn * K_ + row) * C_ + col];
      }
    __syncthreads();
#pragma unroll
    for (int r = 0; r < 6; ++r) {
      const int row = wave * 6 + r;
      float s = 0.f, ss = 0.f;
      for (int c = lane; c < C_; c += 32) {
        float x = shM[row * C_ + c]; s += x; ss += x * x;
      }
#pragma unroll
      for (int off = 16; off >= 1; off >>= 1) {
        s += __shfl_xor(s, off); ss += __shfl_xor(ss, off);
      }
      const float mu  = s * (1.0f / (float)C_);
      const float inv = rsqrtf(ss * (1.0f / (float)C_) - mu * mu + EPS_);
      for (int c = lane; c < C_; c += 32) {
        out[(bn * K_ + row) * C_ + c] = (shM[row * C_ + c] - mu) * inv * lnG[c] + lnB[c];
      }
    }
  }
}

// per-node: LN1(hV+dh) -> FFN -> LN2(h1+ffn) -> *mask  (tiny, scalar f32)
__global__ __launch_bounds__(128, 1) void node_ffn_kernel(
    const float* __restrict__ hV, const float* __restrict__ dh,
    const float* __restrict__ mask,
    const float* __restrict__ n1g, const float* __restrict__ n1b,
    const float* __restrict__ dinW, const float* __restrict__ dinb,
    const float* __restrict__ doutW, const float* __restrict__ doutb,
    const float* __restrict__ n2g, const float* __restrict__ n2b,
    float* __restrict__ outV)
{
  __shared__ float sx[C_];
  __shared__ float sh[C_];
  __shared__ float sred[8];
  const int bn = blockIdx.x;
  const int c  = threadIdx.x;
  const int lane = c & 31, w = c >> 5;

  float x = hV[bn * C_ + c] + dh[bn * C_ + c];

  float s = x, ss = x * x;
#pragma unroll
  for (int off = 16; off >= 1; off >>= 1) { s += __shfl_xor(s, off); ss += __shfl_xor(ss, off); }
  if (lane == 0) { sred[w] = s; sred[4 + w] = ss; }
  __syncthreads();
  float S  = sred[0] + sred[1] + sred[2] + sred[3];
  float SS = sred[4] + sred[5] + sred[6] + sred[7];
  float mu  = S * (1.0f / (float)C_);
  float inv = rsqrtf(SS * (1.0f / (float)C_) - mu * mu + EPS_);
  float h1 = (x - mu) * inv * n1g[c] + n1b[c];

  __syncthreads();
  sx[c] = h1;
  __syncthreads();
  float a1 = dinb[c];
  for (int i = 0; i < C_; ++i) a1 += dinW[c * C_ + i] * sx[i];
  sh[c] = gelu_f(a1);
  __syncthreads();
  float a2 = doutb[c];
  for (int i = 0; i < C_; ++i) a2 += doutW[c * H_ + i] * sh[i];
  float x2 = h1 + a2;

  s = x2; ss = x2 * x2;
#pragma unroll
  for (int off = 16; off >= 1; off >>= 1) { s += __shfl_xor(s, off); ss += __shfl_xor(ss, off); }
  if (lane == 0) { sred[w] = s; sred[4 + w] = ss; }
  __syncthreads();
  S  = sred[0] + sred[1] + sred[2] + sred[3];
  SS = sred[4] + sred[5] + sred[6] + sred[7];
  mu  = S * (1.0f / (float)C_);
  inv = rsqrtf(SS * (1.0f / (float)C_) - mu * mu + EPS_);
  outV[bn * C_ + c] = ((x2 - mu) * inv * n2g[c] + n2b[c]) * mask[bn];
}

extern "C" void kernel_launch(void* const* d_in, const int* in_sizes, int n_in,
                              void* d_out, int out_size, void* d_ws, size_t ws_size,
                              hipStream_t stream) {
  const float* hV      = (const float*)d_in[0];
  const float* hE      = (const float*)d_in[1];
  const int*   idx     = (const int*)  d_in[2];
  const float* mask    = (const float*)d_in[3];
  const float* maskAtt = (const float*)d_in[4];
  const float* w1W  = (const float*)d_in[5];  const float* w1b  = (const float*)d_in[6];
  const float* w2W  = (const float*)d_in[7];  const float* w2b  = (const float*)d_in[8];
  const float* w3W  = (const float*)d_in[9];  const float* w3b  = (const float*)d_in[10];
  const float* n1g  = (const float*)d_in[11]; const float* n1b  = (const float*)d_in[12];
  const float* diW  = (const float*)d_in[13]; const float* dib  = (const float*)d_in[14];
  const float* doW  = (const float*)d_in[15]; const float* dob  = (const float*)d_in[16];
  const float* n2g  = (const float*)d_in[17]; const float* n2b  = (const float*)d_in[18];
  const float* w11W = (const float*)d_in[19]; const float* w11b = (const float*)d_in[20];
  const float* w12W = (const float*)d_in[21]; const float* w12b = (const float*)d_in[22];
  const float* w13W = (const float*)d_in[23]; const float* w13b = (const float*)d_in[24];
  const float* n3g  = (const float*)d_in[25]; const float* n3b  = (const float*)d_in[26];
  (void)in_sizes; (void)n_in; (void)out_size; (void)ws_size;

  float* outV = (float*)d_out;                 // [B*N*C]
  float* outE = outV + B_ * N_ * C_;           // [B*N*K*C]

  // workspace layout
  float*    dh   = (float*)d_ws;               // [B*N*C] f32 scratch (2 MB)
  _Float16* wh   = (_Float16*)((char*)d_ws + (size_t)B_ * N_ * C_ * sizeof(float));
  _Float16* w1h  = wh;                         // H*E   = 49152
  _Float16* w2h  = w1h + H_ * E_;              // H*H   = 16384
  _Float16* w3h  = w2h + H_ * H_;              // C*H   = 16384
  _Float16* w11h = w3h + C_ * H_;
  _Float16* w12h = w11h + H_ * E_;
  _Float16* w13h = w12h + H_ * H_;

  // pre-convert the six edge-MLP weight matrices to f16 (once per launch)
  cvt_w_kernel<<<(H_ * E_ + 255) / 256, 256, 0, stream>>>(w1W,  w1h,  H_ * E_);
  cvt_w_kernel<<<(H_ * H_ + 255) / 256, 256, 0, stream>>>(w2W,  w2h,  H_ * H_);
  cvt_w_kernel<<<(C_ * H_ + 255) / 256, 256, 0, stream>>>(w3W,  w3h,  C_ * H_);
  cvt_w_kernel<<<(H_ * E_ + 255) / 256, 256, 0, stream>>>(w11W, w11h, H_ * E_);
  cvt_w_kernel<<<(H_ * H_ + 255) / 256, 256, 0, stream>>>(w12W, w12h, H_ * H_);
  cvt_w_kernel<<<(C_ * H_ + 255) / 256, 256, 0, stream>>>(w13W, w13h, C_ * H_);

  dim3 grid(B_ * N_);
  edge_mlp_kernel<0><<<grid, 256, 0, stream>>>(
      hV, hE, idx, w1h, w1b, w2h, w2b, w3h, w3b, maskAtt, nullptr, nullptr, dh);
  node_ffn_kernel<<<grid, 128, 0, stream>>>(
      hV, dh, mask, n1g, n1b, diW, dib, doW, dob, n2g, n2b, outV);
  edge_mlp_kernel<1><<<grid, 256, 0, stream>>>(
      outV, hE, idx, w11h, w11b, w12h, w12b, w13h, w13b, nullptr, n3g, n3b, outE);
}
